// rnn_message_passing_global_v3_44495861187273
// MI455X (gfx1250) — matrix operand order repaired and checked by reading, not compiled, not run on hardware
//
#include <hip/hip_runtime.h>
#include <hip/hip_bf16.h>
#include <math.h>

typedef __attribute__((ext_vector_type(16))) _Float16 v16h;
typedef __attribute__((ext_vector_type(8)))  float    v8f;
typedef __attribute__((ext_vector_type(4)))  int      v4i;

#define NW 8  // waves per block in edge kernels

// ---- optional CDNA5 async global->LDS path (guarded; falls back cleanly) ----
#if defined(__has_builtin)
#if __has_builtin(__builtin_amdgcn_global_load_async_to_lds_b128) && \
    __has_builtin(__builtin_amdgcn_s_wait_asynccnt)
#define USE_ASYNC_LDS 1
#endif
#endif
#ifndef USE_ASYNC_LDS
#define USE_ASYNC_LDS 0
#endif

#if USE_ASYNC_LDS
#define GLOBAL_AS __attribute__((address_space(1)))
#define LDS_AS    __attribute__((address_space(3)))
#endif

// ---- fragment loaders (two 16B LDS loads each) -------------------------
// A fragment (16x32 f16, M=lane&15): halves j0..7 at K=c0.., j8..15 at K=c0+16..
__device__ inline v16h ld_frag_a(const _Float16* s) {
  union { v16h v; uint4 q[2]; } u;
  u.q[0] = *(const uint4*)(s);
  u.q[1] = *(const uint4*)(s + 16);
  return u.v;
}
// B fragment (32x16 f16, N=lane&15): halves j0..7 at K=c0.., j8..15 at K=c0+16..
__device__ inline v16h ld_frag_b(const _Float16* s) {
  union { v16h v; uint4 q[2]; } u;
  u.q[0] = *(const uint4*)(s);
  u.q[1] = *(const uint4*)(s + 16);
  return u.v;
}

__device__ inline float sigm(float x) { return 1.0f / (1.0f + __expf(-x)); }

// ======================= node LSTM (scalar, tiny) =======================
__global__ __launch_bounds__(256) void node_lstm_kernel(
    const float* __restrict__ x, const float* __restrict__ u,
    const float* __restrict__ h, const float* __restrict__ c,
    const float* __restrict__ Wih, const float* __restrict__ Whh,
    const float* __restrict__ bih, const float* __restrict__ bhh,
    float* __restrict__ node_hist, float* __restrict__ node_c,
    _Float16* __restrict__ xu_h, int N)
{
  __shared__ float sWih[80 * 37];
  __shared__ float sWhh[80 * 20];
  __shared__ float sb[80];
  for (int i = threadIdx.x; i < 80 * 37; i += 256) sWih[i] = Wih[i];
  for (int i = threadIdx.x; i < 80 * 20; i += 256) sWhh[i] = Whh[i];
  for (int i = threadIdx.x; i < 80; i += 256) sb[i] = bih[i] + bhh[i];
  __syncthreads();
  int i = blockIdx.x * 256 + threadIdx.x;
  if (i >= N) return;

  float xr[37];
#pragma unroll
  for (int k = 0; k < 5; ++k) xr[k] = x[(size_t)i * 5 + k];
#pragma unroll
  for (int k = 0; k < 32; ++k) xr[5 + k] = u[(size_t)i * 32 + k];
  float hr[20];
#pragma unroll
  for (int k = 0; k < 20; ++k) hr[k] = h[(size_t)i * 20 + k];

#pragma unroll
  for (int k = 0; k < 37; ++k) xu_h[(size_t)i * 40 + k] = (_Float16)xr[k];
#pragma unroll
  for (int k = 37; k < 40; ++k) xu_h[(size_t)i * 40 + k] = (_Float16)0.f;

  for (int j = 0; j < 20; ++j) {
    float g4[4];
#pragma unroll
    for (int q = 0; q < 4; ++q) {
      int g = q * 20 + j;
      float a = sb[g];
#pragma unroll
      for (int k = 0; k < 37; ++k) a += xr[k] * sWih[g * 37 + k];
#pragma unroll
      for (int k = 0; k < 20; ++k) a += hr[k] * sWhh[g * 20 + k];
      g4[q] = a;
    }
    float cold = c[(size_t)i * 20 + j];
    float cn = sigm(g4[1]) * cold + sigm(g4[0]) * tanhf(g4[2]);
    float hn = sigm(g4[3]) * tanhf(cn);
    node_hist[(size_t)i * 20 + j] = hn;
    node_c[(size_t)i * 20 + j] = cn;
  }
}

// ======================= edge MLP via WMMA ==============================
// Padded input layout per edge row (K dimension, halves):
//   [0, NF)            : src-node features
//   [NF, SRCP)         : zero padding (SRCP multiple of 8 -> 16B aligned)
//   [SRCP, SRCP+NF)    : dst-node features
//   [SRCP+NF, SRCP+NF+4): edge_attr (overwrites copied zero tail)
//   rest up to KP      : static zeros
// W1 rows are remapped in LDS to match (zero rows at padding positions).
template <int NF, int SRCP, int FS, int KP, int KCH, bool DO_CNT>
__global__ __launch_bounds__(256) void edge_mlp_wmma(
    const _Float16* __restrict__ feat, const int* __restrict__ ei,
    const float* __restrict__ eattr,
    const float* __restrict__ W1, const float* __restrict__ b1,
    const float* __restrict__ W2, const float* __restrict__ b2,
    float* __restrict__ agg, float* __restrict__ cnt, int E, int numBT)
{
  __shared__ __align__(16) _Float16 sW1t[64 * KP];      // [n][k] transposed
  __shared__ __align__(16) _Float16 sW2t[32 * 64];      // [n][k]
  __shared__ __align__(16) _Float16 sIn[NW * 16 * KP];  // per-wave edge rows
  __shared__ __align__(16) _Float16 sH1[NW * 16 * 64];  // per-wave hidden
  __shared__ int sRow[NW * 16];
  __shared__ int sCol[NW * 16];

  const int tid = threadIdx.x;
  const int wave = tid >> 5, lane = tid & 31;
  const int half = lane >> 4, mn = lane & 15;
  constexpr int EOFF = SRCP + NF;   // eattr K position
  constexpr int NCH = SRCP / 8;     // 16B chunks per node row

  // weights -> LDS once per (persistent) block, rows remapped to padded K
  for (int i = tid; i < 64 * KP; i += 256) {
    int n = i / KP, p = i % KP;
    int k = -1;
    if (p < NF) k = p;
    else if (p >= SRCP && p < SRCP + NF + 4) k = p - (SRCP - NF);
    sW1t[i] = (k >= 0) ? (_Float16)W1[k * 64 + n] : (_Float16)0.f;
  }
  for (int i = tid; i < 32 * 64; i += 256) {
    int n = i / 64, k = i % 64;
    sW2t[i] = (_Float16)W2[k * 32 + n];
  }
  // zero sIn once: establishes the static zero tail [EOFF+4, KP)
  for (int i = tid; i < NW * 16 * KP; i += 256) sIn[i] = (_Float16)0.f;
  __syncthreads();

  const v8f vzero = {0.f, 0.f, 0.f, 0.f, 0.f, 0.f, 0.f, 0.f};

  for (int bt = blockIdx.x; bt < numBT; bt += gridDim.x) {
    const int tile0 = (bt * NW + wave) * 16;

    if (lane < 16) {
      int e = tile0 + lane;
      int r = 0, cc = 0;
      if (e < E) { r = ei[e]; cc = ei[(size_t)E + e]; }
      sRow[wave * 16 + lane] = r;
      sCol[wave * 16 + lane] = cc;
    }
    __syncthreads();

    // bulk-gather node feature rows as 16B chunks.
    // Preferred path: CDNA5 async global->LDS (no VGPR round-trip, ASYNCcnt).
    for (int t = lane; t < 16 * 2 * NCH; t += 32) {
      int m = t / (2 * NCH);
      int c = t % (2 * NCH);
      int node = (c < NCH) ? sRow[wave * 16 + m] : sCol[wave * 16 + m];
      int cc2 = (c < NCH) ? c : c - NCH;
      const uint4* src = (const uint4*)(feat + (size_t)node * FS) + cc2;
      uint4* dst = (uint4*)(sIn + (wave * 16 + m) * KP) + c;
#if USE_ASYNC_LDS
      __builtin_amdgcn_global_load_async_to_lds_b128(
          (GLOBAL_AS v4i*)src, (LDS_AS v4i*)dst, 0, 0);
#else
      *dst = *src;
#endif
    }
#if USE_ASYNC_LDS
    __builtin_amdgcn_s_wait_asynccnt(0);
#endif
    __syncthreads();

    // edge attributes: one b128 row load per edge, f32 -> f16 into col tail
    if (lane < 16) {
      int e = tile0 + lane;
      float4 ea = make_float4(0.f, 0.f, 0.f, 0.f);
      if (e < E) ea = *(const float4*)(eattr + (size_t)e * 4);
      _Float16* d = sIn + (wave * 16 + lane) * KP + EOFF;
      d[0] = (_Float16)ea.x;
      d[1] = (_Float16)ea.y;
      d[2] = (_Float16)ea.z;
      d[3] = (_Float16)ea.w;
    }
    __syncthreads();

    // ---- GEMM1: [16 x KP] @ [KP x 64] ----
    v8f acc[4];
#pragma unroll
    for (int t = 0; t < 4; ++t) acc[t] = vzero;

    const _Float16* aBase = sIn + (wave * 16 + mn) * KP;
#pragma unroll
    for (int kc = 0; kc < KCH; ++kc) {
      v16h a = ld_frag_a(aBase + kc * 32 + half * 8);
#pragma unroll
      for (int nt = 0; nt < 4; ++nt) {
        v16h b = ld_frag_b(sW1t + (nt * 16 + mn) * KP + kc * 32 + half * 8);
        acc[nt] = __builtin_amdgcn_wmma_f32_16x16x32_f16(
            false, a, false, b, (short)0, acc[nt], false, false);
      }
    }

    // bias + ReLU, write hidden as f16 (A-layout source for GEMM2)
#pragma unroll
    for (int nt = 0; nt < 4; ++nt) {
#pragma unroll
      for (int r = 0; r < 8; ++r) {
        int m = r + half * 8;
        int n = nt * 16 + mn;
        float v = acc[nt][r] + b1[n];
        v = v > 0.f ? v : 0.f;
        sH1[(wave * 16 + m) * 64 + n] = (_Float16)v;
      }
    }
    __syncthreads();

    // ---- GEMM2: [16 x 64] @ [64 x 32] ----
    v8f acc2[2];
    acc2[0] = vzero; acc2[1] = vzero;
    const _Float16* a2 = sH1 + (wave * 16 + mn) * 64;
#pragma unroll
    for (int kc = 0; kc < 2; ++kc) {
      v16h a = ld_frag_a(a2 + kc * 32 + half * 8);
#pragma unroll
      for (int nt = 0; nt < 2; ++nt) {
        v16h b = ld_frag_b(sW2t + (nt * 16 + mn) * 64 + kc * 32 + half * 8);
        acc2[nt] = __builtin_amdgcn_wmma_f32_16x16x32_f16(
            false, a, false, b, (short)0, acc2[nt], false, false);
      }
    }

    // scatter: agg[col[e]][n] += msg + b2[n]
#pragma unroll
    for (int nt = 0; nt < 2; ++nt) {
#pragma unroll
      for (int r = 0; r < 8; ++r) {
        int m = r + half * 8;
        int e = tile0 + m;
        if (e < E) {
          int n = nt * 16 + mn;
          float v = acc2[nt][r] + b2[n];
          atomicAdd(&agg[(size_t)sCol[wave * 16 + m] * 32 + n], v);
        }
      }
    }
    if (DO_CNT && lane < 16) {
      int e = tile0 + lane;
      if (e < E) atomicAdd(&cnt[sCol[wave * 16 + lane]], 1.0f);
    }
    __syncthreads();  // protect shared state before next iteration
  }
}

// ======================= edge LSTM + full_h build =======================
__global__ __launch_bounds__(256) void edge_lstm_kernel(
    const float* __restrict__ agg1, const float* __restrict__ cnt,
    const float* __restrict__ h, const float* __restrict__ c,
    const float* __restrict__ Wih, const float* __restrict__ Whh,
    const float* __restrict__ bih, const float* __restrict__ bhh,
    const float* __restrict__ node_hist,
    float* __restrict__ edge_hist, float* __restrict__ edge_c,
    _Float16* __restrict__ full_h, int N)
{
  __shared__ float sWih[32 * 32];
  __shared__ float sWhh[32 * 8];
  __shared__ float sb[32];
  for (int i = threadIdx.x; i < 32 * 32; i += 256) sWih[i] = Wih[i];
  for (int i = threadIdx.x; i < 32 * 8; i += 256) sWhh[i] = Whh[i];
  for (int i = threadIdx.x; i < 32; i += 256) sb[i] = bih[i] + bhh[i];
  __syncthreads();
  int i = blockIdx.x * 256 + threadIdx.x;
  if (i >= N) return;

  float inv = 1.0f / fmaxf(cnt[i], 1.0f);
  float ar[32];
#pragma unroll
  for (int k = 0; k < 32; ++k) ar[k] = agg1[(size_t)i * 32 + k] * inv;
  float hr[8];
#pragma unroll
  for (int k = 0; k < 8; ++k) hr[k] = h[(size_t)i * 8 + k];

  float ehr[8];
  for (int j = 0; j < 8; ++j) {
    float g4[4];
#pragma unroll
    for (int q = 0; q < 4; ++q) {
      int g = q * 8 + j;
      float a = sb[g];
#pragma unroll
      for (int k = 0; k < 32; ++k) a += ar[k] * sWih[g * 32 + k];
#pragma unroll
      for (int k = 0; k < 8; ++k) a += hr[k] * sWhh[g * 8 + k];
      g4[q] = a;
    }
    float cold = c[(size_t)i * 8 + j];
    float cn = sigm(g4[1]) * cold + sigm(g4[0]) * tanhf(g4[2]);
    float hn = sigm(g4[3]) * tanhf(cn);
    edge_hist[(size_t)i * 8 + j] = hn;
    edge_c[(size_t)i * 8 + j] = cn;
    ehr[j] = hn;
  }
  // full = [node_hist(20) | edge_hist(8) | pad(4)] as f16 (stride 32)
#pragma unroll
  for (int k = 0; k < 20; ++k) full_h[(size_t)i * 32 + k] = (_Float16)node_hist[(size_t)i * 20 + k];
#pragma unroll
  for (int k = 0; k < 8; ++k) full_h[(size_t)i * 32 + 20 + k] = (_Float16)ehr[k];
#pragma unroll
  for (int k = 28; k < 32; ++k) full_h[(size_t)i * 32 + k] = (_Float16)0.f;
}

// ======================= final node MLP =================================
__global__ __launch_bounds__(256) void node_out_kernel(
    const float* __restrict__ node_hist, const float* __restrict__ edge_hist,
    const float* __restrict__ agg2, const float* __restrict__ cnt,
    const float* __restrict__ Wn1, const float* __restrict__ bn1,
    const float* __restrict__ Wn2, const float* __restrict__ bn2,
    float* __restrict__ out, int N)
{
  __shared__ float sW1[60 * 64];
  __shared__ float sb1[64];
  __shared__ float sW2[64 * 4];
  __shared__ float sb2[4];
  for (int i = threadIdx.x; i < 60 * 64; i += 256) sW1[i] = Wn1[i];
  for (int i = threadIdx.x; i < 64; i += 256) sb1[i] = bn1[i];
  for (int i = threadIdx.x; i < 64 * 4; i += 256) sW2[i] = Wn2[i];
  if (threadIdx.x < 4) sb2[threadIdx.x] = bn2[threadIdx.x];
  __syncthreads();
  int i = blockIdx.x * 256 + threadIdx.x;
  if (i >= N) return;

  float in[60];
#pragma unroll
  for (int k = 0; k < 20; ++k) in[k] = node_hist[(size_t)i * 20 + k];
#pragma unroll
  for (int k = 0; k < 8; ++k) in[20 + k] = edge_hist[(size_t)i * 8 + k];
  float inv = 1.0f / fmaxf(cnt[i], 1.0f);
#pragma unroll
  for (int k = 0; k < 32; ++k) in[28 + k] = agg2[(size_t)i * 32 + k] * inv;

  float o[4] = {sb2[0], sb2[1], sb2[2], sb2[3]};
  for (int n = 0; n < 64; ++n) {
    float hn = sb1[n];
#pragma unroll
    for (int k = 0; k < 60; ++k) hn += in[k] * sW1[k * 64 + n];
    hn = fmaxf(hn, 0.f);
#pragma unroll
    for (int q = 0; q < 4; ++q) o[q] += hn * sW2[n * 4 + q];
  }
#pragma unroll
  for (int q = 0; q < 4; ++q) out[(size_t)i * 4 + q] = o[q];
}

// ======================= launcher =======================================
extern "C" void kernel_launch(void* const* d_in, const int* in_sizes, int n_in,
                              void* d_out, int out_size, void* d_ws, size_t ws_size,
                              hipStream_t stream) {
  const float* x     = (const float*)d_in[0];
  const float* u     = (const float*)d_in[1];
  const int*   ei    = (const int*)d_in[2];
  const float* eattr = (const float*)d_in[3];
  const float* hnh = (const float*)d_in[4];
  const float* hnc = (const float*)d_in[5];
  const float* heh = (const float*)d_in[6];
  const float* hec = (const float*)d_in[7];
  const float* Wih_n = (const float*)d_in[8];
  const float* Whh_n = (const float*)d_in[9];
  const float* bih_n = (const float*)d_in[10];
  const float* bhh_n = (const float*)d_in[11];
  const float* Wih_e = (const float*)d_in[12];
  const float* Whh_e = (const float*)d_in[13];
  const float* bih_e = (const float*)d_in[14];
  const float* bhh_e = (const float*)d_in[15];
  const float* We1 = (const float*)d_in[16]; const float* be1 = (const float*)d_in[17];
  const float* We2 = (const float*)d_in[18]; const float* be2 = (const float*)d_in[19];
  const float* Wo1 = (const float*)d_in[20]; const float* bo1 = (const float*)d_in[21];
  const float* Wo2 = (const float*)d_in[22]; const float* bo2 = (const float*)d_in[23];
  const float* Wn1 = (const float*)d_in[24]; const float* bn1 = (const float*)d_in[25];
  const float* Wn2 = (const float*)d_in[26]; const float* bn2 = (const float*)d_in[27];

  const int N = in_sizes[0] / 5;
  const int E = in_sizes[2] / 2;

  float* out  = (float*)d_out;
  float* o_nh = out + (size_t)4 * N;
  float* o_nc = out + (size_t)24 * N;
  float* o_eh = out + (size_t)44 * N;
  float* o_ec = out + (size_t)52 * N;

  char* ws = (char*)d_ws;
  size_t off = 0;
  _Float16* xu_h = (_Float16*)(ws + off);
  off += (((size_t)N * 40 * 2) + 255) & ~(size_t)255;
  _Float16* full_h = (_Float16*)(ws + off);
  off += (((size_t)N * 32 * 2) + 255) & ~(size_t)255;
  float* agg1 = (float*)(ws + off); off += (size_t)N * 32 * 4;
  float* agg2 = (float*)(ws + off); off += (size_t)N * 32 * 4;
  float* cnt  = (float*)(ws + off); off += (size_t)N * 4;

  // zero accumulators (agg1 | agg2 | cnt are contiguous)
  (void)hipMemsetAsync(agg1, 0, ((size_t)N * 64 + N) * sizeof(float), stream);

  const int nb = (N + 255) / 256;
  const int numBT = (E + NW * 16 - 1) / (NW * 16);  // 128 edges per block-tile
  int eb = numBT < 1024 ? numBT : 1024;             // persistent blocks
  if (eb < 1) eb = 1;

  node_lstm_kernel<<<nb, 256, 0, stream>>>(x, u, hnh, hnc, Wih_n, Whh_n, bih_n, bhh_n,
                                           o_nh, o_nc, xu_h, N);
  // stage 1: NF=37, SRCP=40, FS=40, KP=96, KCH=3
  edge_mlp_wmma<37, 40, 40, 96, 3, true><<<eb, 256, 0, stream>>>(
      xu_h, ei, eattr, We1, be1, We2, be2, agg1, cnt, E, numBT);
  edge_lstm_kernel<<<nb, 256, 0, stream>>>(agg1, cnt, heh, hec, Wih_e, Whh_e, bih_e, bhh_e,
                                           o_nh, o_eh, o_ec, full_h, N);
  // stage 2: NF=28, SRCP=32, FS=32, KP=64, KCH=2
  edge_mlp_wmma<28, 32, 32, 64, 2, false><<<eb, 256, 0, stream>>>(
      full_h, ei, eattr, Wo1, bo1, Wo2, bo2, agg2, cnt, E, numBT);
  node_out_kernel<<<nb, 256, 0, stream>>>(o_nh, o_eh, agg2, cnt, Wn1, bn1, Wn2, bn2, out, N);
}
